// TokenQKAttentionIMU_79216376807585
// MI455X (gfx1250) — matrix operand order repaired and checked by reading, not compiled, not run on hardware
//
#include <hip/hip_runtime.h>

#define T_DIM 4
#define B_DIM 64
#define D_DIM 512
#define V_DIM 256
#define HD    128

typedef __attribute__((ext_vector_type(16))) __bf16 v16bf;
typedef __attribute__((ext_vector_type(8)))  float  v8f;

union AFrag { v16bf bf; uint4 q[2]; };
union BFrag { v16bf bf; uint4 q[2]; };

static __device__ __forceinline__ unsigned short f2bf(float f) {
    unsigned int u = __float_as_uint(f);
    u += 0x7FFFu + ((u >> 16) & 1u);          // round-to-nearest-even
    return (unsigned short)(u >> 16);
}

// pack two fp32 -> one dword holding 2 bf16 (lo = a, hi = b): 2 adds + 1 v_perm_b32
static __device__ __forceinline__ unsigned int pack2bf(float a, float b) {
    unsigned int ax = __float_as_uint(a) + 0x8000u;   // round-half-up into bf16
    unsigned int bx = __float_as_uint(b) + 0x8000u;
    return __builtin_amdgcn_perm(bx, ax, 0x07060302); // D = {bx[31:16], ax[31:16]}
}

static __device__ __forceinline__ v8f wmma_bf16(const v16bf& a, const v16bf& b, const v8f& c) {
    return __builtin_amdgcn_wmma_f32_16x16x32_bf16(false, a, false, b, (short)0, c, false, false);
}

// ---------------- prep: weights fp32 -> bf16 ----------------
__global__ __launch_bounds__(256) void prep_w(const float* __restrict__ wq,
                                              const float* __restrict__ wk,
                                              const float* __restrict__ wp,
                                              unsigned short* __restrict__ dst) {
    int i = blockIdx.x * 256 + threadIdx.x;   // 0 .. 3*512*512-1
    const int N = D_DIM * D_DIM;
    float v;
    if (i < N)          v = wq[i];
    else if (i < 2 * N) v = wk[i - N];
    else                v = wp[i - 2 * N];
    dst[i] = f2bf(v);
}

// ---------------- prep: fold BN into scale/bias ----------------
__global__ __launch_bounds__(256) void prep_bn(
    const float* qg, const float* qb, const float* qm, const float* qv,
    const float* kg, const float* kb, const float* km, const float* kv,
    const float* pg, const float* pb, const float* pm, const float* pv,
    float* __restrict__ coef) {
    int i = blockIdx.x * 256 + threadIdx.x;
    if (i < D_DIM) {
        float inv;
        inv = qg[i] * rsqrtf(qv[i] + 1e-5f); coef[i]        = inv; coef[512 + i]  = qb[i] - qm[i] * inv;
        inv = kg[i] * rsqrtf(kv[i] + 1e-5f); coef[1024 + i] = inv; coef[1536 + i] = kb[i] - km[i] * inv;
        inv = pg[i] * rsqrtf(pv[i] + 1e-5f); coef[2048 + i] = inv; coef[2560 + i] = pb[i] - pm[i] * inv;
    }
}

// ---------------- kernel 1: Q/K GEMM + BN + LIF + head-gate -> bf16 spikes (transposed) ----------------
// 512 threads = 16 waves: waves 0-7 do the Q conv (128 rows = one head), waves 8-15 do the K conv.
__global__ __launch_bounds__(512) void snn_qk_gate(
    const float* __restrict__ x,
    const unsigned short* __restrict__ Wq_bf,
    const unsigned short* __restrict__ Wk_bf,
    const float* __restrict__ coef,
    unsigned short* __restrict__ spkT) {            // [T][B][V][D] bf16
    __shared__ uint4 xs4[64 * 32];                  // 32 KB: [v][d-chunk16B], chunk ^= (v&31)
    __shared__ float qsum[64];
    __shared__ float gspk[64];
    __shared__ float vgate[64];

    const int tid  = threadIdx.x;
    const int wave = tid >> 5;
    const int lane = tid & 31;
    const int ln   = lane & 15;
    const int hf   = lane >> 4;
    const int cw   = __builtin_amdgcn_readfirstlane(wave >> 3);  // 0 = Q conv, 1 = K conv
    const int w8   = wave & 7;

    const int bid   = blockIdx.x;                   // B * H * (V/64) = 1024
    const int b     = bid >> 4;
    const int h     = (bid >> 2) & 3;
    const int vt    = bid & 3;
    const int vbase = vt * 64;
    const int rowb  = h * HD + w8 * 16;             // wave's 16 output rows (inside head h)
    const int orow  = rowb + ln;

    const unsigned short* W = cw ? Wk_bf : Wq_bf;
    const int cbase = cw * 1024;

    if (tid < 64) vgate[tid] = 0.0f;

    __builtin_prefetch(W + (size_t)orow * D_DIM, 0, 0);

    float sc[8], bi[8];
#pragma unroll
    for (int r = 0; r < 8; ++r) {
        int m = rowb + r + 8 * hf;
        sc[r] = coef[cbase + m];
        bi[r] = coef[cbase + 512 + m];
    }

    float vst[4][8];
#pragma unroll
    for (int nt = 0; nt < 4; ++nt)
#pragma unroll
        for (int r = 0; r < 8; ++r) vst[nt][r] = 0.0f;

#pragma unroll 1
    for (int t = 0; t < T_DIM; ++t) {
        v8f acc[4];
#pragma unroll
        for (int nt = 0; nt < 4; ++nt)
#pragma unroll
            for (int r = 0; r < 8; ++r) acc[nt][r] = 0.0f;

        for (int kh = 0; kh < 2; ++kh) {
            __syncthreads();                        // xs4 reuse fence
            {   // stage 256(K) x 64(V) fp32 -> bf16, transposed to [v][d]; 1 block / thread
                const float* xsrc = x + (((size_t)t * B_DIM + b) * D_DIM + kh * 256) * V_DIM + vbase;
                const int vb = tid & 15;            // v-block (4 cols)
                const int db = tid >> 4;            // d-block (8 rows), 0..31
                const float* p0 = xsrc + (size_t)(db * 8) * V_DIM + vb * 4;
#pragma unroll
                for (int p = 0; p < 2; ++p) {       // 4 d-rows at a time -> b64 stores
                    float4 rr[4];
#pragma unroll
                    for (int jr = 0; jr < 4; ++jr)
                        rr[jr] = *(const float4*)(p0 + (size_t)(p * 4 + jr) * V_DIM);
                    const float* ra = (const float*)rr;  // ra[jr*4 + j]
#pragma unroll
                    for (int j = 0; j < 4; ++j) {
                        const int v = vb * 4 + j;
                        uint2 pk;
                        pk.x = pack2bf(ra[0 * 4 + j], ra[1 * 4 + j]);
                        pk.y = pack2bf(ra[2 * 4 + j], ra[3 * 4 + j]);
                        ((uint2*)&xs4[v * 32 + (db ^ (v & 31))])[p] = pk;
                    }
                }
                if (kh == 0 && tid < 64) qsum[tid] = 0.0f;
            }
            __syncthreads();

#pragma unroll 1                                     // keep rolled: avoids hoisting 64 LDS addrs
            for (int ks8 = 0; ks8 < 8; ++ks8) {
                const int kb32 = kh * 256 + ks8 * 32;
                AFrag a;
                const uint4* ap = (const uint4*)(W + (size_t)orow * D_DIM + kb32 + hf * 8);
                a.q[0] = ap[0]; a.q[1] = ap[2];              // K: +0..7 and +16..23 (lane half)
                const int c0 = ks8 * 4 + hf;                 // 16B chunk index in [v][d] row
#pragma unroll
                for (int nt = 0; nt < 4; ++nt) {
                    const int col = nt * 16 + ln;
                    BFrag bf_;
                    bf_.q[0] = xs4[col * 32 + ((c0)     ^ (col & 31))];
                    bf_.q[1] = xs4[col * 32 + ((c0 + 2) ^ (col & 31))];
                    acc[nt] = wmma_bf16(a.bf, bf_.bf, acc[nt]);
                }
            }
        }

        // --- Q waves: BN + LIF + per-column spike sum over the head's 128 rows ---
        if (cw == 0) {
#pragma unroll
            for (int nt = 0; nt < 4; ++nt) {
                float ps = 0.0f;
#pragma unroll
                for (int r = 0; r < 8; ++r) {
                    float y = acc[nt][r] * sc[r] + bi[r];
                    float v = vst[nt][r] + (y - vst[nt][r]) * 0.5f;
                    float s = (v >= 1.0f) ? 1.0f : 0.0f;
                    vst[nt][r] = v * (1.0f - s);
                    ps += s;
                }
                atomicAdd(&qsum[nt * 16 + ln], ps);          // ds_add_f32
            }
        }
        __syncthreads();

        if (tid < 64) {                                      // gate LIF, v_th = 0.5
            float v = vgate[tid] + (qsum[tid] - vgate[tid]) * 0.5f;
            float g = (v >= 0.5f) ? 1.0f : 0.0f;
            vgate[tid] = v * (1.0f - g);
            gspk[tid] = g;
        }
        __syncthreads();

        // --- K waves: BN + LIF, gate, pack 8 row-spikes -> one b128 store (transposed layout) ---
        if (cw == 1) {
            unsigned short* dT = spkT + ((size_t)(t * B_DIM + b) * V_DIM + vbase) * D_DIM;
#pragma unroll
            for (int nt = 0; nt < 4; ++nt) {
                float g = gspk[nt * 16 + ln];
                unsigned int hw[8];
#pragma unroll
                for (int r = 0; r < 8; ++r) {
                    float y = acc[nt][r] * sc[r] + bi[r];
                    float v = vst[nt][r] + (y - vst[nt][r]) * 0.5f;
                    float s = (v >= 1.0f) ? 1.0f : 0.0f;
                    vst[nt][r] = v * (1.0f - s);
                    hw[r] = ((g * s) >= 1.0f) ? 0x3F80u : 0u;  // bf16 1.0 or 0.0
                }
                uint4 pk;
                pk.x = hw[0] | (hw[1] << 16);
                pk.y = hw[2] | (hw[3] << 16);
                pk.z = hw[4] | (hw[5] << 16);
                pk.w = hw[6] | (hw[7] << 16);
                *(uint4*)&dT[(size_t)(nt * 16 + ln) * D_DIM + rowb + 8 * hf] = pk;
            }
        }
    }
}

// ---------------- kernel 2: projection GEMM + BN + LIF -> fp32 spikes ----------------
// 512 threads = 16 waves: wave group 0 covers n-tiles 0-1, group 1 covers n-tiles 2-3.
__global__ __launch_bounds__(512) void snn_proj(
    const unsigned short* __restrict__ spkT,     // [T][B][V][D] bf16
    const unsigned short* __restrict__ Wp_bf,
    const float* __restrict__ coef,
    float* __restrict__ out) {
    __shared__ uint4 xs4[64 * 32];

    const int tid  = threadIdx.x;
    const int wave = tid >> 5;
    const int lane = tid & 31;
    const int ln   = lane & 15;
    const int hf   = lane >> 4;
    const int cw   = __builtin_amdgcn_readfirstlane(wave >> 3);  // n-tile group
    const int w8   = wave & 7;

    const int bid   = blockIdx.x;                // B * (D/128) * (V/64) = 1024
    const int b     = bid >> 4;
    const int ot    = (bid >> 2) & 3;
    const int vt    = bid & 3;
    const int vbase = vt * 64;
    const int rowb  = ot * 128 + w8 * 16;
    const int orow  = rowb + ln;

    __builtin_prefetch(Wp_bf + (size_t)orow * D_DIM, 0, 0);

    float sP[8], bP[8];
#pragma unroll
    for (int r = 0; r < 8; ++r) {
        int m = rowb + r + 8 * hf;
        sP[r] = coef[2048 + m]; bP[r] = coef[2560 + m];
    }

    float vp[2][8];
#pragma unroll
    for (int nt = 0; nt < 2; ++nt)
#pragma unroll
        for (int r = 0; r < 8; ++r) vp[nt][r] = 0.0f;

#pragma unroll 1
    for (int t = 0; t < T_DIM; ++t) {
        v8f acc[2];
#pragma unroll
        for (int nt = 0; nt < 2; ++nt)
#pragma unroll
            for (int r = 0; r < 8; ++r) acc[nt][r] = 0.0f;

        for (int kh = 0; kh < 2; ++kh) {
            __syncthreads();
            {   // pure b128 tile copy: spikes already bf16 and [v][d]-major
                const int chunk = tid & 31;
                const int rv0   = tid >> 5;                  // 0..15
                const uint4* sp4 = (const uint4*)spkT;
                const size_t rowbase = (size_t)(t * B_DIM + b) * V_DIM + vbase;
#pragma unroll
                for (int i = 0; i < 4; ++i) {
                    const int rv = rv0 + 16 * i;
                    uint4 d = sp4[(rowbase + rv) * 64 + kh * 32 + chunk];
                    xs4[rv * 32 + (chunk ^ (rv & 31))] = d;
                }
            }
            __syncthreads();

#pragma unroll 1                                 // keep rolled: avoids hoisting LDS addrs
            for (int ks8 = 0; ks8 < 8; ++ks8) {
                const int kb32 = kh * 256 + ks8 * 32;
                AFrag a;
                const uint4* ap = (const uint4*)(Wp_bf + (size_t)orow * D_DIM + kb32 + hf * 8);
                a.q[0] = ap[0]; a.q[1] = ap[2];
                const int c0 = ks8 * 4 + hf;
#pragma unroll
                for (int nt = 0; nt < 2; ++nt) {
                    const int col = (cw * 2 + nt) * 16 + ln;
                    BFrag bf_;
                    bf_.q[0] = xs4[col * 32 + ((c0)     ^ (col & 31))];
                    bf_.q[1] = xs4[col * 32 + ((c0 + 2) ^ (col & 31))];
                    acc[nt] = wmma_bf16(a.bf, bf_.bf, acc[nt]);
                }
            }
        }

        float* dst = out + (size_t)(t * B_DIM + b) * D_DIM * V_DIM + vbase;
#pragma unroll
        for (int nt = 0; nt < 2; ++nt) {
#pragma unroll
            for (int r = 0; r < 8; ++r) {
                float y = acc[nt][r] * sP[r] + bP[r];
                float v = vp[nt][r] + (y - vp[nt][r]) * 0.5f;
                float s = (v >= 1.0f) ? 1.0f : 0.0f;
                vp[nt][r] = v * (1.0f - s);
                dst[(size_t)(rowb + r + 8 * hf) * V_DIM + (cw * 2 + nt) * 16 + ln] = s;
            }
        }
    }
}

extern "C" void kernel_launch(void* const* d_in, const int* in_sizes, int n_in,
                              void* d_out, int out_size, void* d_ws, size_t ws_size,
                              hipStream_t stream) {
    (void)in_sizes; (void)n_in; (void)out_size; (void)ws_size;

    const float* x  = (const float*)d_in[0];
    const float* Wq = (const float*)d_in[1];
    const float* qg = (const float*)d_in[2];
    const float* qb = (const float*)d_in[3];
    const float* qm = (const float*)d_in[4];
    const float* qv = (const float*)d_in[5];
    const float* Wk = (const float*)d_in[6];
    const float* kg = (const float*)d_in[7];
    const float* kb = (const float*)d_in[8];
    const float* km = (const float*)d_in[9];
    const float* kv = (const float*)d_in[10];
    const float* Wp = (const float*)d_in[11];
    const float* pg = (const float*)d_in[12];
    const float* pb = (const float*)d_in[13];
    const float* pm = (const float*)d_in[14];
    const float* pv = (const float*)d_in[15];

    unsigned short* Wq_bf = (unsigned short*)d_ws;
    unsigned short* Wk_bf = Wq_bf + 512 * 512;
    unsigned short* Wp_bf = Wk_bf + 512 * 512;
    float* coef = (float*)((char*)d_ws + (size_t)3 * 512 * 512 * 2);    // 6*512 floats
    unsigned short* spkT = (unsigned short*)((char*)d_ws + (2u << 20)); // 64 MB bf16 spikes [T][B][V][D]

    hipLaunchKernelGGL(prep_w,      dim3(3 * 512 * 512 / 256), dim3(256), 0, stream, Wq, Wk, Wp, Wq_bf);
    hipLaunchKernelGGL(prep_bn,     dim3(2),    dim3(256), 0, stream,
                       qg, qb, qm, qv, kg, kb, km, kv, pg, pb, pm, pv, coef);
    hipLaunchKernelGGL(snn_qk_gate, dim3(1024), dim3(512), 0, stream, x, Wq_bf, Wk_bf, coef, spkT);
    hipLaunchKernelGGL(snn_proj,    dim3(1024), dim3(512), 0, stream, spkT, Wp_bf, coef, (float*)d_out);
}
